// Head_35364760715568
// MI455X (gfx1250) — matrix operand order repaired and checked by reading, not compiled, not run on hardware
//
#include <hip/hip_runtime.h>

typedef __attribute__((ext_vector_type(16))) _Float16 v16h;
typedef __attribute__((ext_vector_type(8)))  float    v8f;

#define BATCH  4
#define TSEQ   4096
#define DMODEL 1024
#define HSZ    64
#define MROWS  (BATCH * TSEQ)

union AV { v16h v; unsigned u[8]; _Float16 h16[16]; };

__device__ __forceinline__ int a_pair_off(int i, int h) {
  // 16-bit A-matrix 16x32 layout (contiguous, matches B): VGPR i holds K pair
  // lanes 0-15: K = 2*i ; lanes 16-31: +16
  return 2 * i + 16 * h;
}

// ---------------- Stage 0: weight conversion ----------------
// W [D,HS] f32 -> Wt [HS,D] f16 (so GEMM B-operand reads are contiguous)
__global__ void cvt_wT(const float* __restrict__ w, _Float16* __restrict__ wT) {
  int i = blockIdx.x * blockDim.x + threadIdx.x;
  if (i >= DMODEL * HSZ) return;
  int d = i / HSZ, c = i % HSZ;
  wT[c * DMODEL + d] = (_Float16)w[i];
}

// ---------------- Stage 1: fused QKV projection GEMM ----------------
// grid.x = MROWS/16 ; block = 128 (4 waves). Wave w computes the 16x16
// output tile (rows blockIdx.x*16, cols 16w) for q, k and v.
// x is read as f32 and converted to f16 in registers (no extra pass).
// q is pre-scaled by D^-0.5 = 1/32 (fold softmax scale into Q).

__global__ void qkv_gemm(const float* __restrict__ x,
                         const _Float16* __restrict__ wqT,
                         const _Float16* __restrict__ wkT,
                         const _Float16* __restrict__ wvT,
                         _Float16* __restrict__ q16,
                         _Float16* __restrict__ k16,
                         _Float16* __restrict__ v16) {
  const int lane = threadIdx.x & 31;
  const int wave = threadIdx.x >> 5;          // N-tile 0..3
  const int row0 = blockIdx.x * 16;
  const int m = lane & 15;
  const int h = lane >> 4;

  v8f aq = {}, ak = {}, av = {};
  const float* xrow = x + (size_t)(row0 + m) * DMODEL;
  const int n = wave * 16 + m;                 // output column (B operand)
  const _Float16* bqp = wqT + (size_t)n * DMODEL;
  const _Float16* bkp = wkT + (size_t)n * DMODEL;
  const _Float16* bvp = wvT + (size_t)n * DMODEL;

  for (int k0 = 0; k0 < DMODEL; k0 += 32) {
    AV a;
#pragma unroll
    for (int i = 0; i < 8; ++i) {
      const float2 f = *(const float2*)(xrow + k0 + a_pair_off(i, h));
      a.h16[2 * i]     = (_Float16)f.x;
      a.h16[2 * i + 1] = (_Float16)f.y;
    }

    const int kb = k0 + h * 16;               // B: 16 contiguous K halves/lane
    v16h bq = *(const v16h*)(bqp + kb);
    v16h bk = *(const v16h*)(bkp + kb);
    v16h bv = *(const v16h*)(bvp + kb);

    aq = __builtin_amdgcn_wmma_f32_16x16x32_f16(false, a.v, false, bq, (short)0, aq, false, false);
    ak = __builtin_amdgcn_wmma_f32_16x16x32_f16(false, a.v, false, bk, (short)0, ak, false, false);
    av = __builtin_amdgcn_wmma_f32_16x16x32_f16(false, a.v, false, bv, (short)0, av, false, false);
  }

#pragma unroll
  for (int r = 0; r < 8; ++r) {
    const size_t row = (size_t)(row0 + r + 8 * h);
    const int col = wave * 16 + m;
    q16[row * HSZ + col] = (_Float16)(aq[r] * 0.03125f);  // fold 1/sqrt(D)
    k16[row * HSZ + col] = (_Float16)ak[r];
    v16[row * HSZ + col] = (_Float16)av[r];
  }
}

// ---------------- Stage 2: flash attention ----------------
// grid = (TSEQ/64, BATCH); block = 128 (4 waves, 16 query rows each).
// KV tiles of 32 keys staged in LDS; V stored transposed so the P@V
// B-operand is lane-contiguous. Online softmax, fp32 accumulators.

__global__ void flash_attn(const _Float16* __restrict__ q16,
                           const _Float16* __restrict__ k16,
                           const _Float16* __restrict__ v16,
                           float* __restrict__ out) {
  __shared__ __align__(32) _Float16 Klds[32 * 64];   // [key][hdim]
  __shared__ __align__(32) _Float16 Vt[64 * 32];     // [hdim][key] (transposed)
  __shared__ __align__(32) _Float16 Pw[4][16 * 32];  // per-wave P tile [qrow][key]

  const int b    = blockIdx.y;
  const int q0   = blockIdx.x * 64;
  const int lane = threadIdx.x & 31;
  const int wave = threadIdx.x >> 5;
  const int m = lane & 15;
  const int h = lane >> 4;

  const _Float16* qb = q16 + (size_t)b * TSEQ * HSZ;
  const _Float16* kb = k16 + (size_t)b * TSEQ * HSZ;
  const _Float16* vb = v16 + (size_t)b * TSEQ * HSZ;

  // Preload this wave's 16x64 Q tile as two K=32 A-operands.
  AV qa[2];
  {
    const _Float16* qrow = qb + (size_t)(q0 + wave * 16 + m) * HSZ;
#pragma unroll
    for (int c = 0; c < 2; ++c)
#pragma unroll
      for (int i = 0; i < 8; ++i)
        qa[c].u[i] = *(const unsigned*)(qrow + 32 * c + a_pair_off(i, h));
  }

  float mrow[8], lsum[8];
  v8f o[4];
#pragma unroll
  for (int r = 0; r < 8; ++r) { mrow[r] = -1e30f; lsum[r] = 0.0f; }
#pragma unroll
  for (int nc = 0; nc < 4; ++nc)
#pragma unroll
    for (int r = 0; r < 8; ++r) o[nc][r] = 0.0f;

  // Cooperative-load lane mapping: 4 threads per KV row, 16 halves each.
  const int lrow = threadIdx.x >> 2;          // 0..31
  const int lc0  = (threadIdx.x & 3) * 16;    // 0,16,32,48

  for (int kv0 = 0; kv0 < TSEQ; kv0 += 32) {
    // ---- stage K tile and transposed V tile into LDS ----
    {
      const uint4* ks = (const uint4*)(kb + (size_t)(kv0 + lrow) * HSZ + lc0);
      uint4* kd = (uint4*)(Klds + lrow * 64 + lc0);
      kd[0] = ks[0];
      kd[1] = ks[1];
      const _Float16* vs = vb + (size_t)(kv0 + lrow) * HSZ + lc0;
#pragma unroll
      for (int j = 0; j < 16; ++j)
        Vt[(lc0 + j) * 32 + lrow] = vs[j];
    }
    if (kv0 + 32 < TSEQ) {
      __builtin_prefetch(kb + (size_t)(kv0 + 32 + lrow) * HSZ + lc0, 0, 0);
      __builtin_prefetch(vb + (size_t)(kv0 + 32 + lrow) * HSZ + lc0, 0, 0);
    }
    __syncthreads();

    // ---- S = Q @ K^T for two 16-key sub-tiles ----
    v8f s0 = {}, s1 = {};
#pragma unroll
    for (int c = 0; c < 2; ++c) {
      v16h b0 = *(const v16h*)(Klds + m * 64 + 32 * c + 16 * h);
      v16h b1 = *(const v16h*)(Klds + (16 + m) * 64 + 32 * c + 16 * h);
      s0 = __builtin_amdgcn_wmma_f32_16x16x32_f16(false, qa[c].v, false, b0, (short)0, s0, false, false);
      s1 = __builtin_amdgcn_wmma_f32_16x16x32_f16(false, qa[c].v, false, b1, (short)0, s1, false, false);
    }

    // ---- online softmax (rows live in 16-lane halves; reduce via shfl_xor) ----
    float p0[8], p1[8];
#pragma unroll
    for (int r = 0; r < 8; ++r) {
      float mx = fmaxf(s0[r], s1[r]);
      mx = fmaxf(mx, __shfl_xor(mx, 1, 32));
      mx = fmaxf(mx, __shfl_xor(mx, 2, 32));
      mx = fmaxf(mx, __shfl_xor(mx, 4, 32));
      mx = fmaxf(mx, __shfl_xor(mx, 8, 32));
      const float mn = fmaxf(mrow[r], mx);
      const float alpha = __expf(mrow[r] - mn);
      mrow[r] = mn;
      p0[r] = __expf(s0[r] - mn);
      p1[r] = __expf(s1[r] - mn);
      float rs = p0[r] + p1[r];
      rs += __shfl_xor(rs, 1, 32);
      rs += __shfl_xor(rs, 2, 32);
      rs += __shfl_xor(rs, 4, 32);
      rs += __shfl_xor(rs, 8, 32);
      lsum[r] = lsum[r] * alpha + rs;
      o[0][r] *= alpha; o[1][r] *= alpha; o[2][r] *= alpha; o[3][r] *= alpha;
    }

    // ---- restage P (C layout) into LDS, reload in A layout ----
    _Float16* P = Pw[wave];
#pragma unroll
    for (int r = 0; r < 8; ++r) {
      P[(r + 8 * h) * 32 + m]      = (_Float16)p0[r];
      P[(r + 8 * h) * 32 + 16 + m] = (_Float16)p1[r];
    }
    __syncthreads();

    AV pa;
#pragma unroll
    for (int i = 0; i < 8; ++i)
      pa.u[i] = *(const unsigned*)(P + m * 32 + a_pair_off(i, h));

    // ---- O += P @ V across four 16-wide head-dim tiles ----
#pragma unroll
    for (int nc = 0; nc < 4; ++nc) {
      v16h bvv = *(const v16h*)(Vt + (16 * nc + m) * 32 + 16 * h);
      o[nc] = __builtin_amdgcn_wmma_f32_16x16x32_f16(false, pa.v, false, bvv, (short)0, o[nc], false, false);
    }
    __syncthreads();
  }

  // ---- epilogue: normalize and store fp32 output ----
  float* ob = out + ((size_t)b * TSEQ + q0 + wave * 16) * HSZ;
#pragma unroll
  for (int r = 0; r < 8; ++r) {
    const float inv = 1.0f / lsum[r];
#pragma unroll
    for (int nc = 0; nc < 4; ++nc)
      ob[(size_t)(r + 8 * h) * HSZ + 16 * nc + m] = o[nc][r] * inv;
  }
}

// ---------------- host-side launch ----------------

extern "C" void kernel_launch(void* const* d_in, const int* in_sizes, int n_in,
                              void* d_out, int out_size, void* d_ws, size_t ws_size,
                              hipStream_t stream) {
  const float* x  = (const float*)d_in[0];
  const float* Wq = (const float*)d_in[1];
  const float* Wk = (const float*)d_in[2];
  const float* Wv = (const float*)d_in[3];
  float* out = (float*)d_out;

  // workspace layout (f16): wqT | wkT | wvT | q16 | k16 | v16  (~6.7 MB)
  _Float16* wqT = (_Float16*)d_ws;
  _Float16* wkT = wqT + (size_t)DMODEL * HSZ;
  _Float16* wvT = wkT + (size_t)DMODEL * HSZ;
  _Float16* q16 = wvT + (size_t)DMODEL * HSZ;
  _Float16* k16 = q16 + (size_t)MROWS * HSZ;
  _Float16* v16 = k16 + (size_t)MROWS * HSZ;

  // Stage 0: convert + transpose weights to f16
  const int wblocks = (DMODEL * HSZ + 255) / 256;
  cvt_wT<<<wblocks, 256, 0, stream>>>(Wq, wqT);
  cvt_wT<<<wblocks, 256, 0, stream>>>(Wk, wkT);
  cvt_wT<<<wblocks, 256, 0, stream>>>(Wv, wvT);

  // Stage 1: fused QKV projection (1024 blocks x 4 waves), f32->f16 inline
  qkv_gemm<<<MROWS / 16, 128, 0, stream>>>(x, wqT, wkT, wvT, q16, k16, v16);

  // Stage 2: flash attention (64 q-tiles x 4 batches)
  flash_attn<<<dim3(TSEQ / 64, BATCH), 128, 0, stream>>>(q16, k16, v16, out);
}
